// CausalSelfAttention_6356551598607
// MI455X (gfx1250) — matrix operand order repaired and checked by reading, not compiled, not run on hardware
//
#include <hip/hip_runtime.h>
#include <hip/hip_bf16.h>

// ---------------------------------------------------------------------------
// Causal self-attention, B=4 T=2048 C=1024 H=16 D=64, gfx1250.
//   1) convert x -> bf16; transpose+convert weights -> bf16 [N][K]
//   2) qkv GEMM (double-buffered LDS, reg-pipelined global loads)
//   3) flash attention (K/V tiles via TDM tensor_load_to_lds, double buffered)
//   4) projection GEMM -> fp32 out
// ---------------------------------------------------------------------------

typedef __bf16 bf16_t;
typedef __attribute__((ext_vector_type(8)))  __bf16 v8bf;
typedef __attribute__((ext_vector_type(16))) __bf16 v16bf;
typedef __attribute__((ext_vector_type(8)))  float  v8f;
typedef __attribute__((ext_vector_type(4)))  unsigned int u32x4;
typedef __attribute__((ext_vector_type(8)))  int i32x8;
typedef __attribute__((ext_vector_type(4)))  int i32x4;

#define B_DIM 4
#define T_DIM 2048
#define C_DIM 1024
#define H_DIM 16
#define D_DIM 64
#define M_TOK (B_DIM * T_DIM)   /* 8192 */
#define N_QKV (3 * C_DIM)       /* 3072 */
#define LDK   80                /* LDS leading dim: 64 data + 16 pad bf16 (128B + 32B) */

__device__ __forceinline__ bf16_t f2bf(float f) { return (bf16_t)f; }

__device__ __forceinline__ v16bf cat8(v8bf lo, v8bf hi) {
  return __builtin_shufflevector(lo, hi, 0,1,2,3,4,5,6,7,8,9,10,11,12,13,14,15);
}

// A fragment (16x32 MxK): lane holds row (lane&15); lanes 0-15 K{0..7,16..23},
// lanes 16-31 K{8..15,24..31}.
__device__ __forceinline__ v16bf frag_a(const bf16_t* p, int ld, int kbase, int lane) {
  int m = lane & 15;
  int h = (lane >> 4) & 1;
  const bf16_t* r = p + m * ld + kbase + h * 8;
  v8bf lo = *(const v8bf*)(r);
  v8bf hi = *(const v8bf*)(r + 16);
  return cat8(lo, hi);
}

// B fragment (32x16 KxN) from transposed stash Bt[n][k]: lane holds col (lane&15);
// lanes 0-15 K{0..7,16..23}, lanes 16-31 K{8..15,24..31}.
__device__ __forceinline__ v16bf frag_b(const bf16_t* p, int ld, int kbase, int lane) {
  int n = lane & 15;
  int h = (lane >> 4) & 1;
  const bf16_t* r = p + n * ld + kbase + h * 8;
  v8bf lo = *(const v8bf*)(r);
  v8bf hi = *(const v8bf*)(r + 16);
  return cat8(lo, hi);
}

__device__ __forceinline__ v8f wmma_bf16(v16bf a, v16bf b, v8f c) {
  return __builtin_amdgcn_wmma_f32_16x16x32_bf16(false, a, false, b, (short)0, c, false, false);
}

// ---------------------------------------------------------------------------
// TDM descriptor helpers (CDNA5 ISA)
// ---------------------------------------------------------------------------
// group0: [1:0]=count=1, [63:32]=lds_addr, [120:64]=global_addr, [127:126]=type=2
__device__ __forceinline__ u32x4 tdm_g0(unsigned lds_off, unsigned long long gaddr) {
  u32x4 g;
  g.x = 1u;
  g.y = lds_off;
  g.z = (unsigned)(gaddr & 0xFFFFFFFFull);
  g.w = (unsigned)((gaddr >> 32) & 0x1FFFFFFull) | 0x80000000u;
  return g;
}
// group1 for a 2D bf16 tile with LDS padding of 8 DWORDs every 32 DWORDs
// (row = 64 bf16 = 128B data + 32B pad -> LDK=80 elements):
//   data_size=1 (2B), pad_enable=1, pad_interval=4 (32 DW), pad_amount=7 (8 DW)
__device__ __forceinline__ i32x8 tdm_g1_2d(unsigned tdim0, unsigned tdim1,
                                           unsigned tile0, unsigned tile1,
                                           unsigned long long stride0) {
  i32x8 g = {};
  g[0] = (1 << 16) | (1 << 20) | (4 << 22) | (7 << 25);
  g[1] = (int)((tdim0 & 0xFFFFu) << 16);                       // tensor_dim0[15:0]
  g[2] = (int)((tdim0 >> 16) | ((tdim1 & 0xFFFFu) << 16));     // dim0 hi | dim1 lo
  g[3] = (int)((tdim1 >> 16) | ((tile0 & 0xFFFFu) << 16));     // dim1 hi | tile_dim0
  g[4] = (int)(tile1 & 0xFFFFu);                               // tile_dim1, tile_dim2=0
  g[5] = (int)(stride0 & 0xFFFFFFFFull);                       // tensor_dim0_stride lo
  g[6] = (int)((stride0 >> 32) & 0xFFFFull);                   // stride hi | dim1_stride=0
  g[7] = 0;
  return g;
}

// ---------------------------------------------------------------------------
// 1) conversion kernels
// ---------------------------------------------------------------------------
__global__ __launch_bounds__(256) void k_convert(const float* __restrict__ in,
                                                 bf16_t* __restrict__ out, int n) {
  int i = blockIdx.x * 256 + threadIdx.x;
  if (i < n) out[i] = f2bf(in[i]);
}

__global__ __launch_bounds__(256) void k_transpose(const float* __restrict__ in,
                                                   bf16_t* __restrict__ out,
                                                   int K, int N) {
  int i = blockIdx.x * 256 + threadIdx.x;
  if (i < K * N) {
    int k = i / N, n = i % N;
    out[(long)n * K + k] = f2bf(in[i]);
  }
}

// ---------------------------------------------------------------------------
// shared GEMM tile compute: 8 waves, wave -> 64x32, WG tile 128x128, K slab 64
// ---------------------------------------------------------------------------
__device__ __forceinline__ void gemm_tile_compute(const bf16_t* As, const bf16_t* Bs,
                                                  int wr, int wc, int lane,
                                                  v8f acc[4][2]) {
#pragma unroll
  for (int ks = 0; ks < 64; ks += 32) {
    v16bf a[4], bfr[2];
#pragma unroll
    for (int i = 0; i < 4; ++i) a[i] = frag_a(&As[(wr + i * 16) * LDK], LDK, ks, lane);
#pragma unroll
    for (int j = 0; j < 2; ++j) bfr[j] = frag_b(&Bs[(wc + j * 16) * LDK], LDK, ks, lane);
#pragma unroll
    for (int i = 0; i < 4; ++i)
#pragma unroll
      for (int j = 0; j < 2; ++j)
        acc[i][j] = wmma_bf16(a[i], bfr[j], acc[i][j]);
  }
}

// ---------------------------------------------------------------------------
// 2) QKV GEMM: xb[8192,1024] @ w_qkvT -> q[B,H,T,D], k[B,H,T,D], v^T[B,H,D,T]
// grid = 64 (M tiles) * 24 (N tiles); block = 256; double-buffered LDS
// ---------------------------------------------------------------------------
__global__ __launch_bounds__(256) void k_qkv_gemm(const bf16_t* __restrict__ xb,
                                                  const bf16_t* __restrict__ wT,
                                                  bf16_t* __restrict__ qb,
                                                  bf16_t* __restrict__ kb,
                                                  bf16_t* __restrict__ vtb) {
  __shared__ __attribute__((aligned(16))) bf16_t As[2][128 * LDK];
  __shared__ __attribute__((aligned(16))) bf16_t Bs[2][128 * LDK];
  int bid = blockIdx.x;
  int m0 = (bid & 63) * 128;
  int n0 = (bid >> 6) * 128;
  int tid = threadIdx.x;
  int lane = tid & 31;
  int wave = tid >> 5;
  int wr = (wave & 1) * 64;
  int wc = (wave >> 1) * 32;
  int srow = tid >> 3;            // staging row 0..31 (x4)
  int skc  = (tid & 7) * 8;       // staging k col

  v8f acc[4][2] = {};
  v8bf ra[4], rb[4];

#pragma unroll
  for (int c = 0; c < 4; ++c) {
    int row = srow + c * 32;
    ra[c] = *(const v8bf*)&xb[(long)(m0 + row) * C_DIM + skc];
    rb[c] = *(const v8bf*)&wT[(long)(n0 + row) * C_DIM + skc];
  }
#pragma unroll
  for (int c = 0; c < 4; ++c) {
    int row = srow + c * 32;
    *(v8bf*)&As[0][row * LDK + skc] = ra[c];
    *(v8bf*)&Bs[0][row * LDK + skc] = rb[c];
  }
  __syncthreads();

  int buf = 0;
  for (int k0 = 64; k0 < C_DIM; k0 += 64) {
#pragma unroll
    for (int c = 0; c < 4; ++c) {            // prefetch next slab into registers
      int row = srow + c * 32;
      ra[c] = *(const v8bf*)&xb[(long)(m0 + row) * C_DIM + k0 + skc];
      rb[c] = *(const v8bf*)&wT[(long)(n0 + row) * C_DIM + k0 + skc];
    }
    gemm_tile_compute(As[buf], Bs[buf], wr, wc, lane, acc);
#pragma unroll
    for (int c = 0; c < 4; ++c) {
      int row = srow + c * 32;
      *(v8bf*)&As[buf ^ 1][row * LDK + skc] = ra[c];
      *(v8bf*)&Bs[buf ^ 1][row * LDK + skc] = rb[c];
    }
    __syncthreads();
    buf ^= 1;
  }
  gemm_tile_compute(As[buf], Bs[buf], wr, wc, lane, acc);

#pragma unroll
  for (int i = 0; i < 4; ++i) {
#pragma unroll
    for (int j = 0; j < 2; ++j) {
#pragma unroll
      for (int r = 0; r < 8; ++r) {
        int m = m0 + wr + i * 16 + ((lane >> 4) << 3) + r;
        int n = n0 + wc + j * 16 + (lane & 15);
        bf16_t v = f2bf(acc[i][j][r]);
        int part = n >> 10;
        int cc = n & 1023;
        int h = cc >> 6;
        int d = cc & 63;
        int b = m >> 11;
        int t = m & 2047;
        long bh = (long)(b * H_DIM + h);
        if (part == 0)      qb[(bh * T_DIM + t) * D_DIM + d] = v;
        else if (part == 1) kb[(bh * T_DIM + t) * D_DIM + d] = v;
        else                vtb[(bh * D_DIM + d) * T_DIM + t] = v;
      }
    }
  }
}

// ---------------------------------------------------------------------------
// 3) Flash attention: grid = 64 (b*h) * 16 (q tiles of 128); block = 256.
// K/V chunks (64 keys) loaded by the Tensor Data Mover, double buffered.
// ---------------------------------------------------------------------------
__global__ __launch_bounds__(256) void k_attn(const bf16_t* __restrict__ qb,
                                              const bf16_t* __restrict__ kb,
                                              const bf16_t* __restrict__ vtb,
                                              bf16_t* __restrict__ yb) {
  __shared__ __attribute__((aligned(16))) bf16_t Ks[2][64 * LDK];
  __shared__ __attribute__((aligned(16))) bf16_t Vs[2][64 * LDK];
  __shared__ __attribute__((aligned(16))) bf16_t Ps[8 * 16 * LDK];

  int bid = blockIdx.x;
  int qt = bid & 15;
  int bh = bid >> 4;
  int tid = threadIdx.x, lane = tid & 31, wave = tid >> 5;
  int q0 = qt * 128 + wave * 16;

  const bf16_t* Qbase = qb + ((long)bh * T_DIM + q0) * D_DIM;
  const bf16_t* Kbh   = kb + (long)bh * T_DIM * D_DIM;
  const bf16_t* Vbh   = vtb + (long)bh * D_DIM * T_DIM;   // [d][t]

  v16bf aq[2];
  aq[0] = frag_a(Qbase, D_DIM, 0, lane);
  aq[1] = frag_a(Qbase, D_DIM, 32, lane);

  float mrow[8], lrow[8];
#pragma unroll
  for (int r = 0; r < 8; ++r) { mrow[r] = -1e30f; lrow[r] = 0.0f; }
  v8f oacc[4] = {};

  bf16_t* Pl = &Ps[wave * 16 * LDK];
  const int kend = qt * 128 + 128;
  const unsigned long long kgaddr = (unsigned long long)(size_t)Kbh;
  const unsigned long long vgaddr = (unsigned long long)(size_t)Vbh;

  // TDM issue for one 64-key chunk: K tile rows=key (stride D), V^T tile rows=d (stride T)
  auto issue_tdm = [&](int kbase, int b2) {
    unsigned kls = (unsigned)(size_t)&Ks[b2][0];
    unsigned vls = (unsigned)(size_t)&Vs[b2][0];
    i32x4 z4 = {};
    i32x8 z8 = {};
    u32x4 g0k = tdm_g0(kls, kgaddr + (unsigned long long)kbase * D_DIM * 2ull);
    i32x8 g1k = tdm_g1_2d(D_DIM, 64, D_DIM, 64, D_DIM);
    __builtin_amdgcn_tensor_load_to_lds(g0k, g1k, z4, z4, z8, 0);
    u32x4 g0v = tdm_g0(vls, vgaddr + (unsigned long long)kbase * 2ull);
    i32x8 g1v = tdm_g1_2d(D_DIM, 64, D_DIM, 64, T_DIM);
    __builtin_amdgcn_tensor_load_to_lds(g0v, g1v, z4, z4, z8, 0);
  };

  if (wave == 0) issue_tdm(0, 0);

  int buf = 0;
  for (int kbase = 0; kbase < kend; kbase += 64) {
    __builtin_amdgcn_s_wait_tensorcnt(0);   // chunk `kbase` resident (wave0 gates barrier)
    __syncthreads();
    if (wave == 0 && kbase + 64 < kend) issue_tdm(kbase + 64, buf ^ 1);

    const bf16_t* Kc = Ks[buf];
    const bf16_t* Vc = Vs[buf];

    // S = Q K^T (16 x 64 per wave)
    v8f s[4] = {};
#pragma unroll
    for (int j = 0; j < 4; ++j) {
#pragma unroll
      for (int ks = 0; ks < 2; ++ks) {
        v16bf bk = frag_b(&Kc[(j * 16) * LDK], LDK, ks * 32, lane);
        s[j] = wmma_bf16(aq[ks], bk, s[j]);
      }
    }

    // scale, causal mask, chunk row max
    float cmax[8];
#pragma unroll
    for (int r = 0; r < 8; ++r) cmax[r] = -1e30f;
#pragma unroll
    for (int j = 0; j < 4; ++j) {
#pragma unroll
      for (int r = 0; r < 8; ++r) {
        float v = s[j][r] * 0.125f;
        int qrow = q0 + ((lane >> 4) << 3) + r;
        int key  = kbase + j * 16 + (lane & 15);
        v = (key <= qrow) ? v : -1e30f;
        s[j][r] = v;
        cmax[r] = fmaxf(cmax[r], v);
      }
    }
#pragma unroll
    for (int mk = 1; mk < 16; mk <<= 1)
#pragma unroll
      for (int r = 0; r < 8; ++r)
        cmax[r] = fmaxf(cmax[r], __shfl_xor(cmax[r], mk, 32));

    float alpha[8], csum[8];
#pragma unroll
    for (int r = 0; r < 8; ++r) {
      float mn = fmaxf(mrow[r], cmax[r]);
      alpha[r] = __expf(mrow[r] - mn);
      mrow[r] = mn;
      csum[r] = 0.0f;
    }
#pragma unroll
    for (int j = 0; j < 4; ++j) {
#pragma unroll
      for (int r = 0; r < 8; ++r) {
        float p = __expf(s[j][r] - mrow[r]);
        csum[r] += p;
        Pl[(((lane >> 4) << 3) + r) * LDK + j * 16 + (lane & 15)] = f2bf(p);
      }
    }
#pragma unroll
    for (int mk = 1; mk < 16; mk <<= 1)
#pragma unroll
      for (int r = 0; r < 8; ++r)
        csum[r] += __shfl_xor(csum[r], mk, 32);
#pragma unroll
    for (int r = 0; r < 8; ++r) lrow[r] = lrow[r] * alpha[r] + csum[r];

#pragma unroll
    for (int j = 0; j < 4; ++j)
#pragma unroll
      for (int r = 0; r < 8; ++r) oacc[j][r] *= alpha[r];
#pragma unroll
    for (int j = 0; j < 4; ++j) {
#pragma unroll
      for (int ks = 0; ks < 2; ++ks) {
        v16bf ap = frag_a(Pl, LDK, ks * 32, lane);
        v16bf bv = frag_b(&Vc[(j * 16) * LDK], LDK, ks * 32, lane);
        oacc[j] = wmma_bf16(ap, bv, oacc[j]);
      }
    }
    buf ^= 1;
  }

  int b = bh >> 4;
  int h = bh & 15;
#pragma unroll
  for (int j = 0; j < 4; ++j) {
#pragma unroll
    for (int r = 0; r < 8; ++r) {
      int t = q0 + ((lane >> 4) << 3) + r;
      int d = j * 16 + (lane & 15);
      float val = oacc[j][r] / lrow[r];
      yb[((long)(b * T_DIM + t)) * C_DIM + h * D_DIM + d] = f2bf(val);
    }
  }
}

// ---------------------------------------------------------------------------
// 4) projection GEMM: yb[8192,1024] @ w_projT -> out fp32 (double buffered)
// ---------------------------------------------------------------------------
__global__ __launch_bounds__(256) void k_proj_gemm(const bf16_t* __restrict__ yb,
                                                   const bf16_t* __restrict__ wT,
                                                   float* __restrict__ out) {
  __shared__ __attribute__((aligned(16))) bf16_t As[2][128 * LDK];
  __shared__ __attribute__((aligned(16))) bf16_t Bs[2][128 * LDK];
  int bid = blockIdx.x;
  int m0 = (bid & 63) * 128;
  int n0 = (bid >> 6) * 128;
  int tid = threadIdx.x;
  int lane = tid & 31;
  int wave = tid >> 5;
  int wr = (wave & 1) * 64;
  int wc = (wave >> 1) * 32;
  int srow = tid >> 3;
  int skc  = (tid & 7) * 8;

  v8f acc[4][2] = {};
  v8bf ra[4], rb[4];

#pragma unroll
  for (int c = 0; c < 4; ++c) {
    int row = srow + c * 32;
    ra[c] = *(const v8bf*)&yb[(long)(m0 + row) * C_DIM + skc];
    rb[c] = *(const v8bf*)&wT[(long)(n0 + row) * C_DIM + skc];
  }
#pragma unroll
  for (int c = 0; c < 4; ++c) {
    int row = srow + c * 32;
    *(v8bf*)&As[0][row * LDK + skc] = ra[c];
    *(v8bf*)&Bs[0][row * LDK + skc] = rb[c];
  }
  __syncthreads();

  int buf = 0;
  for (int k0 = 64; k0 < C_DIM; k0 += 64) {
#pragma unroll
    for (int c = 0; c < 4; ++c) {
      int row = srow + c * 32;
      ra[c] = *(const v8bf*)&yb[(long)(m0 + row) * C_DIM + k0 + skc];
      rb[c] = *(const v8bf*)&wT[(long)(n0 + row) * C_DIM + k0 + skc];
    }
    gemm_tile_compute(As[buf], Bs[buf], wr, wc, lane, acc);
#pragma unroll
    for (int c = 0; c < 4; ++c) {
      int row = srow + c * 32;
      *(v8bf*)&As[buf ^ 1][row * LDK + skc] = ra[c];
      *(v8bf*)&Bs[buf ^ 1][row * LDK + skc] = rb[c];
    }
    __syncthreads();
    buf ^= 1;
  }
  gemm_tile_compute(As[buf], Bs[buf], wr, wc, lane, acc);

#pragma unroll
  for (int i = 0; i < 4; ++i) {
#pragma unroll
    for (int j = 0; j < 2; ++j) {
#pragma unroll
      for (int r = 0; r < 8; ++r) {
        int m = m0 + wr + i * 16 + ((lane >> 4) << 3) + r;
        int n = n0 + wc + j * 16 + (lane & 15);
        out[(long)m * C_DIM + n] = acc[i][j][r];
      }
    }
  }
}

// ---------------------------------------------------------------------------
extern "C" void kernel_launch(void* const* d_in, const int* in_sizes, int n_in,
                              void* d_out, int out_size, void* d_ws, size_t ws_size,
                              hipStream_t stream) {
  const float* x      = (const float*)d_in[0];
  const float* w_qkv  = (const float*)d_in[1];
  const float* w_proj = (const float*)d_in[2];
  float* out = (float*)d_out;

  char* ws = (char*)d_ws;
  const size_t szX   = (size_t)M_TOK * C_DIM * sizeof(bf16_t);
  const size_t szWQ  = (size_t)N_QKV * C_DIM * sizeof(bf16_t);
  const size_t szWP  = (size_t)C_DIM * C_DIM * sizeof(bf16_t);
  const size_t szQKV = (size_t)M_TOK * C_DIM * sizeof(bf16_t);

  bf16_t* xb     = (bf16_t*)(ws);                 ws += szX;
  bf16_t* wqkvT  = (bf16_t*)(ws);                 ws += szWQ;
  bf16_t* wprojT = (bf16_t*)(ws);                 ws += szWP;
  bf16_t* qb     = (bf16_t*)(ws);                 ws += szQKV;
  bf16_t* kb     = (bf16_t*)(ws);                 ws += szQKV;
  bf16_t* vtb    = (bf16_t*)(ws);                 ws += szQKV;
  bf16_t* yb     = (bf16_t*)(ws);                 ws += szQKV;
  (void)ws_size; (void)in_sizes; (void)n_in; (void)out_size;

  {
    int n = M_TOK * C_DIM;
    k_convert<<<(n + 255) / 256, 256, 0, stream>>>(x, xb, n);
  }
  {
    int n = C_DIM * N_QKV;
    k_transpose<<<(n + 255) / 256, 256, 0, stream>>>(w_qkv, wqkvT, C_DIM, N_QKV);
  }
  {
    int n = C_DIM * C_DIM;
    k_transpose<<<(n + 255) / 256, 256, 0, stream>>>(w_proj, wprojT, C_DIM, C_DIM);
  }

  k_qkv_gemm<<<(M_TOK / 128) * (N_QKV / 128), 256, 0, stream>>>(xb, wqkvT, qb, kb, vtb);
  k_attn<<<B_DIM * H_DIM * (T_DIM / 128), 256, 0, stream>>>(qb, kb, vtb, yb);
  k_proj_gemm<<<(M_TOK / 128) * (C_DIM / 128), 256, 0, stream>>>(yb, wprojT, out);
}
